// TimesFmAttention_42941083025642
// MI455X (gfx1250) — compile-verified, matrix operand order checked by reading, not run on hardware
//
#include <hip/hip_runtime.h>
#include <math.h>

// ---------------------------------------------------------------------------
// TimesFM attention on gfx1250 (MI455X): bf16 WMMA pipeline with async
// global->LDS tile DMA (ASYNCcnt) and double-buffered LDS tiles.
//   B=4, S=2048, D=1280, H=16, HD=80 (padded to 96)
// Stages: prep(convert+scale) -> QKV GEMM -> flash attention -> O-proj GEMM
// ---------------------------------------------------------------------------

typedef __attribute__((ext_vector_type(16))) __bf16 v16bf;
typedef __attribute__((ext_vector_type(8)))  float  v8f;
typedef __attribute__((ext_vector_type(4)))  int    v4i;

union FragBF { v16bf v; unsigned int u[8]; };
union PackBF { unsigned int u; __bf16 h[2]; };

// ---- CDNA5 async global->LDS copy (tracked by ASYNCcnt) -------------------
// Builtin signature (from clang diagnostic): (v4i AS1*, v4i AS3*, Ii, Ii)
#if __has_builtin(__builtin_amdgcn_global_load_async_to_lds_b128)
#define HAVE_ASYNC_LDS 1
__device__ __forceinline__ void async_b128(void* lds, const void* g) {
  void* gp = const_cast<void*>(g);
  __builtin_amdgcn_global_load_async_to_lds_b128(
      (__attribute__((address_space(1))) v4i*)gp,
      (__attribute__((address_space(3))) v4i*)lds, 0, 0);
}
#else
__device__ __forceinline__ void async_b128(void* lds, const void* g) {
  *(uint4*)lds = *(const uint4*)g;  // synchronous fallback via VGPRs
}
#endif

#if __has_builtin(__builtin_amdgcn_s_wait_asynccnt)
#define WAIT_ASYNC(n) __builtin_amdgcn_s_wait_asynccnt(n)
#else
#define WAIT_ASYNC(n) asm volatile("s_wait_asynccnt %0" ::"i"(n) : "memory")
#endif

// Dword index within a 32-bf16 K-chunk for WMMA A/B operand layout:
// lanes 0-15 (half=0): K pairs {0..7},{16..23}; lanes 16-31 (half=1): {8..15},{24..31}
__device__ __forceinline__ int kb2(int v, int half) {
  return (v & 3) + half * 4 + ((v >> 2) << 3);
}

__device__ __forceinline__ v8f wmma_bf16(v16bf a, v16bf b, v8f c) {
  // 8 args: (neg_a, A, neg_b, B, c_mod, C, reuse_a, reuse_b)
  return __builtin_amdgcn_wmma_f32_16x16x32_bf16(false, a, false, b, (short)0, c,
                                                 false, false);
}

// ---------------------------------------------------------------------------
// Stage 0: f32->bf16 conversions, qscale = softplus(s)*log2e/sqrt(80),
//          zero the HD padding (dims 80..95) of Q/K/V buffers.
// ---------------------------------------------------------------------------
__global__ void timesfm_prep_kernel(const float* __restrict__ hid,
                                    const float* __restrict__ qkvw,
                                    const float* __restrict__ ow,
                                    const float* __restrict__ scaling,
                                    __bf16* __restrict__ hid_bf,
                                    __bf16* __restrict__ qkvw_bf,
                                    __bf16* __restrict__ ow_bf,
                                    float* __restrict__ qscale,
                                    __bf16* __restrict__ qb,
                                    __bf16* __restrict__ kb,
                                    __bf16* __restrict__ vb) {
  long t = (long)blockIdx.x * blockDim.x + threadIdx.x;
  long stride = (long)gridDim.x * blockDim.x;
  for (long i = t; i < 10485760L; i += stride) hid_bf[i]  = (__bf16)hid[i];
  for (long i = t; i < 4915200L;  i += stride) qkvw_bf[i] = (__bf16)qkvw[i];
  for (long i = t; i < 1638400L;  i += stride) ow_bf[i]   = (__bf16)ow[i];
  // zero pad dims 80..95 for all 64*2048 (bh, s) rows of q/k/v
  for (long i = t; i < 2097152L; i += stride) {
    long row = i >> 4;
    int  d   = 80 + (int)(i & 15);
    long off = row * 96 + d;
    qb[off] = (__bf16)0.f;
    kb[off] = (__bf16)0.f;
    vb[off] = (__bf16)0.f;
  }
  if (t < 80) {
    float x  = scaling[t];
    float sp = x > 0.f ? x + log1pf(__expf(-x)) : log1pf(__expf(x));
    qscale[t] = sp * (1.442695041f / 8.94427191f);  // log2e / sqrt(80)
  }
}

// ---------------------------------------------------------------------------
// Tiled bf16 GEMM: C[M,N] = A[M,K] * B[N,K]^T  (both row-major, bf16)
// Block tile 128x128x32, 8 waves, each wave = 32(M) x 64(N) = 2x4 WMMA tiles.
// Double-buffered LDS with async global->LDS prefetch of the next K-tile.
// mode 0: QKV epilogue (bias, per-dim Q scale, scatter to padded q/k/v bf16)
// mode 1: O-proj epilogue (bias, f32 output)
// ---------------------------------------------------------------------------
__global__ __launch_bounds__(256)
void timesfm_gemm_bf16_kernel(const unsigned int* __restrict__ A_u,
                              const unsigned int* __restrict__ B_u,
                              int Kd /* K in dwords (=K/2) */, int mode,
                              const float* __restrict__ bias,
                              const float* __restrict__ qscale,
                              __bf16* __restrict__ qb, __bf16* __restrict__ kbuf,
                              __bf16* __restrict__ vb, float* __restrict__ out) {
  __shared__ unsigned int As_u[2][128 * 16];  // [buf][row][16 dw] = 128x32 bf16
  __shared__ unsigned int Bs_u[2][128 * 16];

  const int tid  = threadIdx.x;
  const int lane = tid & 31;
  const int w    = tid >> 5;
  const int wy   = w >> 1, wx = w & 1;
  const int nl   = lane & 15, half = lane >> 4;
  const long M0  = (long)blockIdx.x * 128;
  const long N0  = (long)blockIdx.y * 128;
  const int nk   = Kd / 16;  // K tiles of 32 bf16

  // issue async copies for one 128x32 A tile + 128x32 B tile (4 b128/thread)
  auto issue_tile = [&](int kt, int buf) {
#pragma unroll
    for (int j = 0; j < 2; ++j) {
      int id  = j * 256 + tid;           // b128 unit: 512 per matrix tile
      int row = id >> 2, c4 = (id & 3) * 4;
      async_b128(&As_u[buf][row * 16 + c4], &A_u[(M0 + row) * Kd + kt * 16 + c4]);
      async_b128(&Bs_u[buf][row * 16 + c4], &B_u[(N0 + row) * Kd + kt * 16 + c4]);
    }
  };

  v8f acc[2][4] = {};

  issue_tile(0, 0);
  for (int kt = 0; kt < nk; ++kt) {
    __syncthreads();  // previous compute done everywhere; buf[(kt+1)&1] free
    if (kt + 1 < nk) {
      issue_tile(kt + 1, (kt + 1) & 1);
      WAIT_ASYNC(4);  // in-order completion: first 4 = current tile landed
    } else {
      WAIT_ASYNC(0);
    }
    __syncthreads();  // current tile visible to all waves

    const unsigned int* Asb = As_u[kt & 1];
    const unsigned int* Bsb = Bs_u[kt & 1];
    FragBF af[2], bfg[4];
#pragma unroll
    for (int mi = 0; mi < 2; ++mi)
#pragma unroll
      for (int v = 0; v < 8; ++v)
        af[mi].u[v] = Asb[(wy * 32 + mi * 16 + nl) * 16 + kb2(v, half)];
#pragma unroll
    for (int ni = 0; ni < 4; ++ni)
#pragma unroll
      for (int v = 0; v < 8; ++v)
        bfg[ni].u[v] = Bsb[(wx * 64 + ni * 16 + nl) * 16 + kb2(v, half)];
#pragma unroll
    for (int mi = 0; mi < 2; ++mi)
#pragma unroll
      for (int ni = 0; ni < 4; ++ni)
        acc[mi][ni] = wmma_bf16(af[mi].v, bfg[ni].v, acc[mi][ni]);
  }

  // Epilogue: C layout row = r + 8*half, col = lane&15 within each 16x16 tile
#pragma unroll
  for (int mi = 0; mi < 2; ++mi) {
#pragma unroll
    for (int ni = 0; ni < 4; ++ni) {
#pragma unroll
      for (int r = 0; r < 8; ++r) {
        long row = M0 + wy * 32 + mi * 16 + r + 8 * half;
        long col = N0 + wx * 64 + ni * 16 + nl;
        float val = acc[mi][ni][r] + bias[col];
        if (mode == 0) {
          int sel = (int)(col / 1280);
          int wi  = (int)(col - (long)sel * 1280);
          int h   = wi / 80, d = wi - h * 80;
          if (sel == 0) val *= qscale[d];
          long b = row >> 11, s = row & 2047;
          long off = ((b * 16 + h) * 2048 + s) * 96 + d;
          __bf16* dst = (sel == 0) ? qb : (sel == 1) ? kbuf : vb;
          dst[off] = (__bf16)val;
        } else {
          out[row * 1280 + col] = val;
        }
      }
    }
  }
}

// ---------------------------------------------------------------------------
// Flash attention: one workgroup = one (b,h) x 128-query block, 8 waves,
// each wave owns 16 query rows. Key tiles of 64, causal. HD padded to 96.
// K tiles double-buffered with async global->LDS prefetch; V manually
// transposed into LDS (needed for the PV B-operand layout).
// ---------------------------------------------------------------------------
__global__ __launch_bounds__(256)
void timesfm_flash_kernel(const unsigned int* __restrict__ q_u,
                          const unsigned int* __restrict__ k_u,
                          const unsigned int* __restrict__ v_u,
                          __bf16* __restrict__ attn_bf) {
  __shared__ unsigned int Ks_u[2][64 * 48];   // K tile [key][96 dims] bf16, x2
  __shared__ unsigned int Vt_u[96 * 32];      // V tile [dim][64 keys] bf16 (transposed)
  __shared__ unsigned int Ps_u[8 * 16 * 32];  // P stage [wave][16 rows][64 cols] bf16
  __bf16* Vt_h = (__bf16*)Vt_u;
  __bf16* Ps_h = (__bf16*)Ps_u;

  const int tid  = threadIdx.x;
  const int lane = tid & 31;
  const int w    = tid >> 5;
  const int nl   = lane & 15, half = lane >> 4;
  const int qblk = blockIdx.x;
  const long bh  = blockIdx.y;          // b*16 + h
  const int qr0  = qblk * 128 + w * 16; // first query row of this wave
  const long seq_base = bh * 2048;

  // issue async copy of one 64x96 K tile (3 b128/thread)
  auto issue_k = [&](int it, int buf) {
    int kt = it * 64;
#pragma unroll
    for (int j = 0; j < 3; ++j) {
      int id  = j * 256 + tid;            // b128 unit: 768 per tile
      int row = id / 12, c4 = (id % 12) * 4;
      async_b128(&Ks_u[buf][row * 48 + c4],
                 &k_u[(seq_base + kt + row) * 48 + c4]);
    }
  };

  // Q fragments: 16 rows x 96 dims = 3 chunks of 32 (A-operand layout)
  FragBF qf[3];
#pragma unroll
  for (int kc = 0; kc < 3; ++kc)
#pragma unroll
    for (int v = 0; v < 8; ++v)
      qf[kc].u[v] = q_u[(seq_base + qr0 + nl) * 48 + kc * 16 + kb2(v, half)];

  v8f oacc[6] = {};
  float m_run[8], l_run[8];
#pragma unroll
  for (int r = 0; r < 8; ++r) { m_run[r] = -1e30f; l_run[r] = 0.f; }

  const int nkt = 2 * (qblk + 1);  // causal: keys up to end of query block
  issue_k(0, 0);
  for (int it = 0; it < nkt; ++it) {
    const int kt = it * 64;
    __syncthreads();  // previous compute done; Vt/Ps free; Ks[(it+1)&1] free

    // V tile: load row-major, store transposed [dim][key] into LDS
#pragma unroll
    for (int j = 0; j < 12; ++j) {
      int dw  = j * 256 + tid;
      int row = dw / 48, cd = dw - row * 48;
      PackBF pv;
      pv.u = v_u[(seq_base + kt + row) * 48 + cd];
      Vt_h[(cd * 2 + 0) * 64 + row] = pv.h[0];
      Vt_h[(cd * 2 + 1) * 64 + row] = pv.h[1];
    }

    if (it + 1 < nkt) {
      issue_k(it + 1, (it + 1) & 1);
      WAIT_ASYNC(3);  // in-order: first 3 = current K tile landed
    } else {
      WAIT_ASYNC(0);
    }
    __syncthreads();  // K tile + transposed V visible to all waves

    const unsigned int* Kb = Ks_u[it & 1];

    // S = Q @ K^T : [16 x 64] (4 n-tiles x 3 k-chunks = 12 WMMA)
    v8f sc[4] = {};
#pragma unroll
    for (int kc = 0; kc < 3; ++kc) {
      FragBF kfrag[4];
#pragma unroll
      for (int nt = 0; nt < 4; ++nt)
#pragma unroll
        for (int v = 0; v < 8; ++v)
          kfrag[nt].u[v] = Kb[(nt * 16 + nl) * 48 + kc * 16 + kb2(v, half)];
#pragma unroll
      for (int nt = 0; nt < 4; ++nt)
        sc[nt] = wmma_bf16(qf[kc].v, kfrag[nt].v, sc[nt]);
    }

    // causal mask + online softmax (row = r + 8*half, col = kt + nt*16 + nl)
#pragma unroll
    for (int r = 0; r < 8; ++r) {
      int rowg = qr0 + r + 8 * half;
      float rmax = -1e30f;
#pragma unroll
      for (int nt = 0; nt < 4; ++nt) {
        int colg = kt + nt * 16 + nl;
        float x = sc[nt][r];
        if (colg > rowg) x = -1e30f;
        sc[nt][r] = x;
        rmax = fmaxf(rmax, x);
      }
#pragma unroll
      for (int off = 1; off < 16; off <<= 1)
        rmax = fmaxf(rmax, __shfl_xor(rmax, off, 32));
      float mnew = fmaxf(m_run[r], rmax);
      float corr = __expf(m_run[r] - mnew);
      m_run[r] = mnew;
      float psum = 0.f;
#pragma unroll
      for (int nt = 0; nt < 4; ++nt) {
        float p = __expf(sc[nt][r] - mnew);
        sc[nt][r] = p;
        psum += p;
      }
#pragma unroll
      for (int off = 1; off < 16; off <<= 1)
        psum += __shfl_xor(psum, off, 32);
      l_run[r] = l_run[r] * corr + psum;
#pragma unroll
      for (int dt = 0; dt < 6; ++dt) oacc[dt][r] *= corr;
    }

    // stage P to LDS as bf16 row-major [16 x 64] (per-wave region)
#pragma unroll
    for (int nt = 0; nt < 4; ++nt)
#pragma unroll
      for (int r = 0; r < 8; ++r)
        Ps_h[(w * 16 + r + 8 * half) * 64 + nt * 16 + nl] = (__bf16)sc[nt][r];

    // O += P @ V : [16 x 96] (6 d-tiles x 2 k-chunks = 12 WMMA)
#pragma unroll
    for (int kc2 = 0; kc2 < 2; ++kc2) {
      FragBF pf;
#pragma unroll
      for (int v = 0; v < 8; ++v)
        pf.u[v] = Ps_u[(w * 16 + nl) * 32 + kc2 * 16 + kb2(v, half)];
#pragma unroll
      for (int dt = 0; dt < 6; ++dt) {
        FragBF vf;
#pragma unroll
        for (int v = 0; v < 8; ++v)
          vf.u[v] = Vt_u[(dt * 16 + nl) * 32 + kc2 * 16 + kb2(v, half)];
        oacc[dt] = wmma_bf16(pf.v, vf.v, oacc[dt]);
      }
    }
  }

  // epilogue: normalize and write [b, s, h*80 + d] (drop padded dims 80..95)
  const long b = bh >> 4;
  const int h  = (int)(bh & 15);
#pragma unroll
  for (int dt = 0; dt < 6; ++dt) {
    int d = dt * 16 + nl;
    if (d >= 80) continue;
#pragma unroll
    for (int r = 0; r < 8; ++r) {
      long s = qr0 + r + 8 * half;
      attn_bf[(b * 2048 + s) * 1280 + h * 80 + d] = (__bf16)(oacc[dt][r] / l_run[r]);
    }
  }
}

// ---------------------------------------------------------------------------
// Host launch
// ---------------------------------------------------------------------------
extern "C" void kernel_launch(void* const* d_in, const int* in_sizes, int n_in,
                              void* d_out, int out_size, void* d_ws, size_t ws_size,
                              hipStream_t stream) {
  const float* hid     = (const float*)d_in[0];
  // d_in[1] = attention_mask (pure causal; reproduced analytically)
  const float* scaling = (const float*)d_in[2];
  const float* qkv_w   = (const float*)d_in[3];
  const float* qkv_b   = (const float*)d_in[4];
  const float* o_w     = (const float*)d_in[5];
  const float* o_b     = (const float*)d_in[6];
  float* out = (float*)d_out;

  char* p = (char*)d_ws;
  auto carve = [&](size_t bytes) -> char* {
    char* r = p;
    p += (bytes + 255) & ~(size_t)255;
    return r;
  };
  __bf16* hid_bf  = (__bf16*)carve(8192L * 1280 * 2);
  __bf16* qkvw_bf = (__bf16*)carve(3840L * 1280 * 2);
  __bf16* ow_bf   = (__bf16*)carve(1280L * 1280 * 2);
  __bf16* qb      = (__bf16*)carve(64L * 2048 * 96 * 2);
  __bf16* kb      = (__bf16*)carve(64L * 2048 * 96 * 2);
  __bf16* vb      = (__bf16*)carve(64L * 2048 * 96 * 2);
  __bf16* attn_bf = (__bf16*)carve(8192L * 1280 * 2);
  float*  qscale  = (float*)carve(512);

  timesfm_prep_kernel<<<2048, 256, 0, stream>>>(hid, qkv_w, o_w, scaling, hid_bf,
                                                qkvw_bf, ow_bf, qscale, qb, kb, vb);

  // QKV: [8192 x 1280] @ [3840 x 1280]^T -> scatter into q/k/v
  timesfm_gemm_bf16_kernel<<<dim3(64, 30), 256, 0, stream>>>(
      (const unsigned int*)hid_bf, (const unsigned int*)qkvw_bf, 640, 0, qkv_b,
      qscale, qb, kb, vb, nullptr);

  // Flash attention: 16 query blocks x 64 (b,h) pairs
  timesfm_flash_kernel<<<dim3(16, 64), 256, 0, stream>>>(
      (const unsigned int*)qb, (const unsigned int*)kb, (const unsigned int*)vb,
      attn_bf);

  // O-proj: [8192 x 1280] @ [1280 x 1280]^T + bias -> f32 out
  timesfm_gemm_bf16_kernel<<<dim3(64, 10), 256, 0, stream>>>(
      (const unsigned int*)attn_bf, (const unsigned int*)ow_bf, 640, 1, o_b,
      qscale, nullptr, nullptr, nullptr, out);
}
